// NextGenerationHyperGNN_56624848831216
// MI455X (gfx1250) — matrix operand order repaired and checked by reading, not compiled, not run on hardware
//
#include <hip/hip_runtime.h>
#include <hip/hip_bf16.h>
#include <math.h>

typedef __attribute__((ext_vector_type(16))) _Float16 v16h;
typedef __attribute__((ext_vector_type(8)))  float    v8f;

static constexpr int kN = 96;    // nodes
static constexpr int kD = 64;    // input dim
static constexpr int kH = 256;   // hidden dim

__device__ __forceinline__ float sigmoidf_(float x) {
    return 1.0f / (1.0f + __expf(-x));
}

// ---------------------------------------------------------------------------
// Kernel A: the five tiny projections X@W  ->  pi, pj, ta, tb, tc  [96 x 256]
// laid out consecutively in workspace. One thread per output element; each
// 256-thread block covers one (matrix, row) so W reads are fully coalesced.
// ---------------------------------------------------------------------------
__global__ void proj_kernel(const float* __restrict__ X,
                            const float* __restrict__ W1,
                            const float* __restrict__ V1,
                            float* __restrict__ ws) {
    int t     = blockIdx.x * blockDim.x + threadIdx.x;   // 0 .. 5*96*256-1
    int which = t / (kN * kH);
    int rem   = t % (kN * kH);
    int r     = rem / kH;
    int h     = rem % kH;
    const float* W;
    switch (which) {
        case 0:  W = W1;            break;  // pi : W1[0:64]
        case 1:  W = W1 + 64 * kH;  break;  // pj : W1[64:128]
        case 2:  W = V1;            break;  // ta : V1[0:64]
        case 3:  W = V1 + 64 * kH;  break;  // tb : V1[64:128]
        default: W = V1 + 128 * kH; break;  // tc : V1[128:192]
    }
    const float* xr = X + r * kD;
    float acc = 0.0f;
    #pragma unroll 4
    for (int d = 0; d < kD; ++d) acc = fmaf(xr[d], W[d * kH + h], acc);
    ws[t] = acc;
}

// ---------------------------------------------------------------------------
// Kernel A2: pack W2 [256,128] (f32, row-major) into f16 B-fragments in the
// contiguous k order this file uses for V_WMMA_F32_16X16X32_F16 (same as A):
//   B is 32x16 (KxN); lanes 0-15 hold N=lane, K=kchunk*32 + 0..15,
//   lanes 16-31 hold N=lane-16, K=kchunk*32 + 16..31 (16 halfs per lane).
// Fragment f = kchunk*8 + ntile; 512 halfs (= 1 KB) each, 64 fragments.
// pair_kernel then loads a fragment with one aligned 32B read per lane.
// ---------------------------------------------------------------------------
__global__ void packW2_kernel(const float* __restrict__ W2,
                              _Float16* __restrict__ w2h) {
    int f    = blockIdx.x;         // 0..63
    int lane = threadIdx.x;        // 0..31
    int kc   = f >> 3;
    int nt   = f & 7;
    int col  = nt * 16 + (lane & 15);
    int kofs = kc * 32 + (lane >> 4) * 16;
    v16h hv;
    #pragma unroll
    for (int e = 0; e < 16; ++e)
        hv[e] = (_Float16)W2[(kofs + e) * 128 + col];
    *(v16h*)(w2h + f * 512 + lane * 16) = hv;
}

// ---------------------------------------------------------------------------
// Kernel B: pair path. One wave32 per tile of 16 pair rows (576 tiles).
//   h1 = relu(pi[i] + pj[j] + b1)                      (built into A fragment)
//   h2 = relu(h1 @ W2 + b2)        via 8x8 = 64 v_wmma_f32_16x16x32_f16
//   causal = sigmoid(h2 @ W3 + b3) fused epilogue with shfl_xor reduction
// A-fragment lane layout (16-bit A 16x32): row M = lane&15;
//   elements e: K = kbase + ((lane>>4)<<4) + e.
// C/D layout: lane holds col N = lane&15, rows M = v + 8*(lane>>4), v=0..7.
// ---------------------------------------------------------------------------
__global__ void pair_kernel(const float* __restrict__ pi,
                            const float* __restrict__ pj,
                            const float* __restrict__ b1,
                            const _Float16* __restrict__ w2h,
                            const float* __restrict__ b2,
                            const float* __restrict__ W3,
                            const float* __restrict__ b3,
                            float* __restrict__ causal) {
    const int tile = blockIdx.x;        // 0..575
    const int lane = threadIdx.x;       // 0..31
    const int r    = lane & 15;
    const int grp  = lane >> 4;

    const int p = tile * 16 + r;        // pair index for this A row
    const int i = p / kN;
    const int j = p % kN;
    const float* pir = pi + i * kH;
    const float* pjr = pj + j * kH;

    v8f acc[8];
    #pragma unroll
    for (int nt = 0; nt < 8; ++nt)
        #pragma unroll
        for (int v = 0; v < 8; ++v) acc[nt][v] = 0.0f;

    #pragma unroll
    for (int kc = 0; kc < 8; ++kc) {
        const int kbase = kc * 32;
        v16h a;
        #pragma unroll
        for (int e = 0; e < 16; ++e) {
            int k   = kbase + (grp << 4) + e;
            float v = pir[k] + pjr[k] + b1[k];
            a[e]    = (_Float16)fmaxf(v, 0.0f);
        }
        #pragma unroll
        for (int nt = 0; nt < 8; ++nt) {
            v16h b = *(const v16h*)(w2h + (kc * 8 + nt) * 512 + lane * 16);
            acc[nt] = __builtin_amdgcn_wmma_f32_16x16x32_f16(
                /*neg_a=*/false, a, /*neg_b=*/false, b,
                /*c_mod=*/(short)0, acc[nt],
                /*reuse_a=*/false, /*reuse_b=*/false);
        }
    }

    // Epilogue: relu(h2 + b2) @ W3, reduced over the 128 columns.
    float partial[8];
    #pragma unroll
    for (int v = 0; v < 8; ++v) partial[v] = 0.0f;
    #pragma unroll
    for (int nt = 0; nt < 8; ++nt) {
        const int   col = nt * 16 + r;
        const float b2c = b2[col];
        const float w3c = W3[col];
        #pragma unroll
        for (int v = 0; v < 8; ++v)
            partial[v] = fmaf(fmaxf(acc[nt][v] + b2c, 0.0f), w3c, partial[v]);
    }
    // Sum across the 16 lanes sharing a row group (masks stay within halves).
    #pragma unroll
    for (int v = 0; v < 8; ++v) {
        float s = partial[v];
        s += __shfl_xor(s, 1, 32);
        s += __shfl_xor(s, 2, 32);
        s += __shfl_xor(s, 4, 32);
        s += __shfl_xor(s, 8, 32);
        partial[v] = s;
    }
    if (r == 0) {
        const float bias3 = b3[0];
        #pragma unroll
        for (int v = 0; v < 8; ++v) {
            int row = v + 8 * grp;
            int pr  = tile * 16 + row;
            int ii  = pr / kN;
            int jj  = pr % kN;
            causal[pr] = (ii == jj) ? 0.0f : sigmoidf_(partial[v] + bias3);
        }
    }
}

// ---------------------------------------------------------------------------
// Kernel C: triplet path (not expressible as GEMM due to the inner relu).
// One wave per (i,j); lane owns 8 h-values of s = ta_i + tb_j + c1 and of V2
// in registers; streams tc rows (L2-resident, 96 KB) and does a full-wave
// shuffle reduction per k.  Block = 8 waves covering 8 consecutive j.
// ---------------------------------------------------------------------------
__global__ void triplet_kernel(const float* __restrict__ ta,
                               const float* __restrict__ tb,
                               const float* __restrict__ tc,
                               const float* __restrict__ c1,
                               const float* __restrict__ V2,
                               const float* __restrict__ c2,
                               float* __restrict__ conf) {
    const int lane = threadIdx.x & 31;
    const int w    = threadIdx.x >> 5;          // 0..7
    const int i    = blockIdx.x / 12;
    const int j    = (blockIdx.x % 12) * 8 + w; // 0..95

    const int h0 = lane * 8;
    float s[8], v2r[8];
    #pragma unroll
    for (int e = 0; e < 8; ++e) {
        const int h = h0 + e;
        s[e]   = ta[i * kH + h] + tb[j * kH + h] + c1[h];
        v2r[e] = V2[h];
    }
    const float c2v   = c2[0];
    float* outrow     = conf + (i * kN + j) * kN;

    for (int k = 0; k < kN; ++k) {
        const float4* t4 = (const float4*)(tc + k * kH + h0);
        const float4 t0 = t4[0];
        const float4 t1 = t4[1];
        const float tv[8] = {t0.x, t0.y, t0.z, t0.w, t1.x, t1.y, t1.z, t1.w};
        float acc = 0.0f;
        #pragma unroll
        for (int e = 0; e < 8; ++e)
            acc = fmaf(fmaxf(s[e] + tv[e], 0.0f), v2r[e], acc);
        acc += __shfl_xor(acc, 1, 32);
        acc += __shfl_xor(acc, 2, 32);
        acc += __shfl_xor(acc, 4, 32);
        acc += __shfl_xor(acc, 8, 32);
        acc += __shfl_xor(acc, 16, 32);
        if (lane == 0) {
            const bool distinct = (i != j) && (j != k) && (i != k);
            outrow[k] = distinct ? sigmoidf_(acc + c2v) : 0.0f;
        }
    }
}

// ---------------------------------------------------------------------------
extern "C" void kernel_launch(void* const* d_in, const int* in_sizes, int n_in,
                              void* d_out, int out_size, void* d_ws, size_t ws_size,
                              hipStream_t stream) {
    (void)in_sizes; (void)n_in; (void)out_size; (void)ws_size;

    const float* X  = (const float*)d_in[0];
    // d_in[1] = edge_index (int) — unused by the reference computation.
    const float* W1 = (const float*)d_in[2];
    const float* b1 = (const float*)d_in[3];
    const float* W2 = (const float*)d_in[4];
    const float* b2 = (const float*)d_in[5];
    const float* W3 = (const float*)d_in[6];
    const float* b3 = (const float*)d_in[7];
    const float* V1 = (const float*)d_in[8];
    const float* c1 = (const float*)d_in[9];
    const float* V2 = (const float*)d_in[10];
    const float* c2 = (const float*)d_in[11];

    // Workspace layout (floats): pi, pj, ta, tb, tc each [96*256], then the
    // packed f16 W2 fragments (64 KB). Total ~557 KB.
    float*    ws  = (float*)d_ws;
    float*    pi  = ws;
    float*    pj  = ws + 1 * kN * kH;
    float*    ta  = ws + 2 * kN * kH;
    float*    tb  = ws + 3 * kN * kH;
    float*    tc  = ws + 4 * kN * kH;
    _Float16* w2h = (_Float16*)(ws + 5 * kN * kH);   // 32-byte aligned offset

    float* causal = (float*)d_out;          // [96*96]
    float* conf   = causal + kN * kN;       // [96*96*96]

    // 5 * 96 * 256 = 122880 outputs, 480 blocks of 256.
    proj_kernel<<<dim3(480), dim3(256), 0, stream>>>(X, W1, V1, ws);
    packW2_kernel<<<dim3(64), dim3(32), 0, stream>>>(W2, w2h);
    pair_kernel<<<dim3(576), dim3(32), 0, stream>>>(pi, pj, b1, w2h, b2, W3, b3, causal);
    triplet_kernel<<<dim3(kN * 12), dim3(256), 0, stream>>>(ta, tb, tc, c1, V2, c2, conf);
}
